// CAPE_Decoder_80341658239446
// MI455X (gfx1250) — compile-verified, matrix operand order checked by reading, not run on hardware
//
#include <hip/hip_runtime.h>
#include <cstdint>
#include <cstddef>

// ---------------------------------------------------------------------------
// CDNA5 (gfx1250) implementation of the CAPE decoder.
// Dense per-node GEMMs: fp32 activations/weights are converted to bf16 ONCE
// (RNE), W is pre-swizzled into the per-lane WMMA B-fragment layout, and the
// GEMM inner loop is pure b128 loads + v_wmma_f32_16x16x32_bf16.
// Sparse graph ops (propagate / barycentric pool) are gather/scatter kernels.
// ---------------------------------------------------------------------------

typedef __attribute__((ext_vector_type(16))) __bf16    v16bf;
typedef __attribute__((ext_vector_type(8)))  float     v8f;
typedef __attribute__((ext_vector_type(8)))  unsigned  v8u;

// round-to-nearest-even fp32 -> bf16 kept in bits [31:16]
__device__ __forceinline__ unsigned rnehi(float f) {
    unsigned u = __builtin_bit_cast(unsigned, f);
    return u + 0x7fffu + ((u >> 16) & 1u);
}
// pack two fp32 into one dword of bf16 pair {hi,lo} with a single v_perm_b32
__device__ __forceinline__ unsigned pack_bf(float lo, float hi) {
    return __builtin_amdgcn_perm(rnehi(hi), rnehi(lo), 0x07060302u);
}

// fuse bits: 1 = +bias[col], 2 = relu, 4 = leaky-relu(0.2), 8 = += res[idx]
#define FUSE_BIAS  1
#define FUSE_RELU  2
#define FUSE_LRELU 4
#define FUSE_RES   8

// fp32 -> packed bf16 pairs (row-major), n2 = element_count/2 (always even here)
__global__ void cvt_bf16(const float* __restrict__ x, unsigned* __restrict__ y, size_t n2)
{
    size_t stride = (size_t)gridDim.x * blockDim.x;
    for (size_t i = (size_t)blockIdx.x * blockDim.x + threadIdx.x; i < n2; i += stride)
        y[i] = pack_bf(x[2 * i], x[2 * i + 1]);
}

// pre-swizzle W[K,N] fp32 into WMMA B fragments:
// flat dword idx = (((tn*(K/32) + ks)*32) + lane)*8 + p ; pair (2p,2p+1)
// holds W[k,n],W[k+1,n] with k = ks*32 + 8*(lane>=16) + e + 8*(e>=8), e=2p,
// n = tn*16 + (lane&15).
__global__ void pack_w_bf16(const float* __restrict__ W, unsigned* __restrict__ Wp,
                            int N, int K)
{
    const int ntN = N >> 4, nks = K >> 5;
    size_t total = (size_t)ntN * nks * 32 * 8;
    size_t stride = (size_t)gridDim.x * blockDim.x;
    for (size_t i = (size_t)blockIdx.x * blockDim.x + threadIdx.x; i < total; i += stride) {
        const int p = (int)(i & 7);
        const int l = (int)((i >> 3) & 31);
        const size_t t = i >> 8;
        const int ks = (int)(t % nks);
        const int tn = (int)(t / nks);
        const int e = 2 * p;
        const int k = ks * 32 + ((l >> 4) << 3) + e + ((e & 8) ? 8 : 0);
        const int n = tn * 16 + (l & 15);
        Wp[i] = pack_bf(W[(size_t)k * N + n], W[(size_t)(k + 1) * N + n]);
    }
}

// C[M,N] = epilogue( A[M,K] @ W[K,N] ), A = bf16 pairs row-major, Wp = packed
// B fragments, C fp32. M%16==0, N%16==0, K%32==0. Grid covers ngM groups of
// MT M-tiles starting at m-tile tmBase; no divergence -> EXEC all-ones.
template <int FUSE, int MT>
__global__ __launch_bounds__(256)
void wmma_gemm(const unsigned* __restrict__ Abf, const unsigned* __restrict__ Wp,
               const float* __restrict__ bias, const float* res,
               float* C, int N, int K, int ngM, int tmBase)
{
    const int lane = threadIdx.x & 31;
    const int ntN  = N >> 4;
    const int grp  = blockIdx.x * (blockDim.x >> 5) + (threadIdx.x >> 5);
    if (grp >= ngM * ntN) return;               // wave-uniform
    const int gm  = grp / ntN;
    const int tn  = grp - gm * ntN;
    const int tm0 = tmBase + gm * MT;

    const int half   = lane >> 4;               // 0: lanes 0-15, 1: lanes 16-31
    const int rl     = lane & 15;
    const int kShift = half << 3;               // +8 K offset for upper lanes
    const int nks    = K >> 5;
    const int Kd     = K >> 1;                  // row stride of Abf in dwords

    v8f acc[MT];
#pragma unroll
    for (int j = 0; j < MT; ++j)
        acc[j] = (v8f){0.f, 0.f, 0.f, 0.f, 0.f, 0.f, 0.f, 0.f};

    const unsigned* Bp = Wp + (size_t)tn * nks * 256 + lane * 8;
    const unsigned* Ap[MT];
#pragma unroll
    for (int j = 0; j < MT; ++j)
        Ap[j] = Abf + (size_t)((tm0 + j) * 16 + rl) * Kd + (kShift >> 1);

    for (int ks = 0; ks < nks; ++ks) {
        // B fragment: 8 contiguous aligned dwords per lane
        const uint4 b0 = *(const uint4*)(Bp);
        const uint4 b1 = *(const uint4*)(Bp + 4);
        Bp += 256;
        const v16bf b = __builtin_bit_cast(
            v16bf, (v8u){b0.x, b0.y, b0.z, b0.w, b1.x, b1.y, b1.z, b1.w});
#pragma unroll
        for (int j = 0; j < MT; ++j) {
            // A fragment: K-runs [kb,kb+7] and [kb+16,kb+23] = 2 aligned b128
            const uint4 a0 = *(const uint4*)(Ap[j]);
            const uint4 a1 = *(const uint4*)(Ap[j] + 8);
            Ap[j] += 16;                         // advance 32 K elements
            const v16bf a = __builtin_bit_cast(
                v16bf, (v8u){a0.x, a0.y, a0.z, a0.w, a1.x, a1.y, a1.z, a1.w});
            acc[j] = __builtin_amdgcn_wmma_f32_16x16x32_bf16(
                false, a, false, b, (short)0, acc[j], false, false);
        }
    }

    // C/D layout: lane<16 -> rows 0..7, lane>=16 -> rows 8..15 ; col = lane&15
    const int col = tn * 16 + rl;
    float bv = 0.f;
    if constexpr (FUSE & FUSE_BIAS) bv = bias[col];
#pragma unroll
    for (int j = 0; j < MT; ++j) {
        const int rbase = (tm0 + j) * 16 + (half << 3);
#pragma unroll
        for (int r = 0; r < 8; ++r) {
            const size_t idx = (size_t)(rbase + r) * N + col;
            float v = acc[j][r] + bv;
            if constexpr (FUSE & FUSE_RES)   v += res[idx];
            if constexpr (FUSE & FUSE_RELU)  v = v > 0.f ? v : 0.f;
            if constexpr (FUSE & FUSE_LRELU) v = v > 0.f ? v : 0.2f * v;
            C[idx] = v;
        }
    }
}

// tiny dense layer (pose MLP): out[M,N] = act(A[M,K]@W[K,N] + b)
__global__ void dense_small(const float* __restrict__ A, const float* __restrict__ W,
                            const float* __restrict__ bias, float* out,
                            int M, int K, int N, int lrelu)
{
    int idx = blockIdx.x * blockDim.x + threadIdx.x;
    if (idx >= M * N) return;
    int r = idx / N, c = idx - r * N;
    float s = bias[c];
    for (int k = 0; k < K; ++k) s += A[r * K + k] * W[k * N + c];
    if (lrelu) s = s > 0.f ? s : 0.2f * s;
    out[idx] = s;
}

// out[b,n,0:C]=h[b,n,:] ; out[b,n,C:C+32]=y1[b,:]
__global__ void concat_cond(const float* __restrict__ h, const float* __restrict__ y1,
                            float* out, int Nn, int C, int bs)
{
    const int Ct = C + 32;
    size_t total = (size_t)bs * Nn * Ct;
    size_t stride = (size_t)gridDim.x * blockDim.x;
    for (size_t i = (size_t)blockIdx.x * blockDim.x + threadIdx.x; i < total; i += stride) {
        int c = (int)(i % Ct);
        size_t bn = i / Ct;
        int n = (int)(bn % Nn);
        int b = (int)(bn / Nn);
        out[i] = (c < C) ? h[((size_t)b * Nn + n) * C + c] : y1[b * 32 + (c - C)];
    }
}

// barycentric upsample (gather): y[b,r,c] = sum_{t<3} val[3r+t]*x[b,col[3r+t],c]
__global__ void pool_up(const float* __restrict__ x, float* y,
                        const int* __restrict__ pcol, const float* __restrict__ pval,
                        int n_out, int n_in, int C, int bs)
{
    size_t total = (size_t)bs * n_out * C;
    size_t stride = (size_t)gridDim.x * blockDim.x;
    for (size_t i = (size_t)blockIdx.x * blockDim.x + threadIdx.x; i < total; i += stride) {
        int c = (int)(i % C);
        size_t br = i / C;
        int r = (int)(br % n_out);
        int b = (int)(br / n_out);
        float acc = 0.f;
#pragma unroll
        for (int t = 0; t < 3; ++t) {
            int e = 3 * r + t;
            acc += pval[e] * x[((size_t)b * n_in + pcol[e]) * C + c];
        }
        y[i] = acc;
    }
}

__global__ void zero_buf(float* p, size_t n)
{
    size_t stride = (size_t)gridDim.x * blockDim.x;
    for (size_t i = (size_t)blockIdx.x * blockDim.x + threadIdx.x; i < n; i += stride)
        p[i] = 0.f;
}

// graph propagate (scatter-add): y[b,row[e],c] += norm[e]*x[b,col[e],c]
__global__ void scatter_prop(const float* __restrict__ x, float* y,
                             const int* __restrict__ row, const int* __restrict__ col,
                             const float* __restrict__ norm, int E, int Nn, int C, int bs)
{
    size_t total = (size_t)bs * E * C;
    size_t stride = (size_t)gridDim.x * blockDim.x;
    for (size_t i = (size_t)blockIdx.x * blockDim.x + threadIdx.x; i < total; i += stride) {
        int c = (int)(i % C);
        size_t be = i / C;
        int e = (int)(be % E);
        int b = (int)(be / E);
        float msg = norm[e] * x[((size_t)b * Nn + col[e]) * C + c];
        atomicAdd(&y[((size_t)b * Nn + row[e]) * C + c], msg);
    }
}

// GroupNorm(32 groups over (N, C/32)) + ReLU ; one block per (batch, group)
__global__ __launch_bounds__(256)
void groupnorm_relu(const float* __restrict__ x, float* y,
                    const float* __restrict__ gamma, const float* __restrict__ beta,
                    int Nn, int C)
{
    const int Cg  = C >> 5;               // C/32
    const int b   = blockIdx.x >> 5;
    const int grp = blockIdx.x & 31;
    const int total = Nn * Cg;

    float s = 0.f, ss = 0.f;
    for (int i = threadIdx.x; i < total; i += blockDim.x) {
        int n = i / Cg, j = i - (i / Cg) * Cg;
        float v = x[((size_t)b * Nn + n) * C + grp * Cg + j];
        s += v; ss += v * v;
    }
    // wave32 reduction then cross-wave via LDS
    for (int o = 16; o; o >>= 1) { s += __shfl_xor(s, o); ss += __shfl_xor(ss, o); }
    __shared__ float sm[16];
    const int wid = threadIdx.x >> 5, ln = threadIdx.x & 31;
    if (ln == 0) { sm[wid] = s; sm[8 + wid] = ss; }
    __syncthreads();
    if (threadIdx.x == 0) {
        float S = 0.f, SS = 0.f;
        const int nw = blockDim.x >> 5;
        for (int w = 0; w < nw; ++w) { S += sm[w]; SS += sm[8 + w]; }
        float mean = S / (float)total;
        float var  = SS / (float)total - mean * mean;
        sm[0] = mean;
        sm[1] = rsqrtf(var + 1e-5f);
    }
    __syncthreads();
    const float mean = sm[0], rstd = sm[1];
    for (int i = threadIdx.x; i < total; i += blockDim.x) {
        int n = i / Cg, j = i - (i / Cg) * Cg;
        int c = grp * Cg + j;
        size_t id = ((size_t)b * Nn + n) * C + c;
        float v = (x[id] - mean) * rstd * gamma[c] + beta[c];
        y[id] = v > 0.f ? v : 0.f;
    }
}

// final K=2 Cheb conv to 3 channels + out_bias
__global__ void out_head(const float* __restrict__ h, const float* __restrict__ ph,
                         const float* __restrict__ w, const float* __restrict__ ob,
                         float* out, int bs, int Nn)
{
    size_t total = (size_t)bs * Nn;
    size_t stride = (size_t)gridDim.x * blockDim.x;
    const float* w0 = w;
    const float* w1 = w + 96 * 3;
    for (size_t i = (size_t)blockIdx.x * blockDim.x + threadIdx.x; i < total; i += stride) {
        int n = (int)(i % Nn);
        float o0 = 0.f, o1 = 0.f, o2 = 0.f;
        const float* hp = h  + i * 96;
        const float* pp = ph + i * 96;
#pragma unroll 4
        for (int c = 0; c < 96; ++c) {
            float hv = hp[c], pv = pp[c];
            o0 += hv * w0[c * 3 + 0] + pv * w1[c * 3 + 0];
            o1 += hv * w0[c * 3 + 1] + pv * w1[c * 3 + 1];
            o2 += hv * w0[c * 3 + 2] + pv * w1[c * 3 + 2];
        }
        out[i * 3 + 0] = o0 + ob[n * 3 + 0];
        out[i * 3 + 1] = o1 + ob[n * 3 + 1];
        out[i * 3 + 2] = o2 + ob[n * 3 + 2];
    }
}

// ---------------------------------------------------------------------------
// host-side sequencing
// ---------------------------------------------------------------------------

// input flattening (depth-first, dict insertion order of setup_inputs()):
enum {
    IN_X = 0, IN_POSE = 1,
    P_PFC1W = 2, P_PFC1B = 3, P_PFC2W = 4, P_PFC2B = 5,
    P_FC1W = 6, P_FC1B = 7, P_CONV1W = 8,
    P_BLOCKS = 9,           // + i*10 : c1,c2,c3,c4,gn1g,gn1b,gn2g,gn2b,gn3g,gn3b
    P_CONVOUTW = 89, P_OUTBIAS = 90,
    G_LEVELS = 91,          // + L*3 : row,col,norm  (L=0..8, N=nodes[L])
    G_UPS = 118             // + u*4 : row,col,val,n_out (u=0..7, n_out=nodes[u])
};

static inline int gsgrid(size_t n, int block = 256, int cap = 1 << 20) {
    size_t g = (n + block - 1) / block;
    if (g > (size_t)cap) g = cap;
    if (g < 1) g = 1;
    return (int)g;
}

template <int FUSE>
static void gemm_launch(const float* A, const float* W, const float* bias,
                        const float* res, float* C, int M, int N, int K,
                        unsigned* abuf, unsigned* wbuf, hipStream_t stream)
{
    // stage A and W as bf16 once (A row-major pairs, W pre-swizzled fragments)
    cvt_bf16<<<gsgrid((size_t)M * K / 2), 256, 0, stream>>>(A, abuf, (size_t)M * K / 2);
    pack_w_bf16<<<gsgrid((size_t)N * K / 2), 256, 0, stream>>>(W, wbuf, N, K);

    const int ntM = M >> 4, ntN = N >> 4;
    const int ng4 = ntM >> 2;                // full groups of 4 M-tiles
    const int tail = ntM & 3;
    if (ng4) {
        const int grid = (ng4 * ntN + 7) / 8;
        wmma_gemm<FUSE, 4><<<grid, 256, 0, stream>>>(abuf, wbuf, bias, res, C, N, K, ng4, 0);
    }
    if (tail) {
        const int grid = (tail * ntN + 7) / 8;
        wmma_gemm<FUSE, 1><<<grid, 256, 0, stream>>>(abuf, wbuf, bias, res, C, N, K,
                                                     tail, ng4 << 2);
    }
}

extern "C" void kernel_launch(void* const* d_in, const int* in_sizes, int n_in,
                              void* d_out, int out_size, void* d_ws, size_t ws_size,
                              hipStream_t stream)
{
    (void)in_sizes; (void)n_in; (void)out_size; (void)ws_size;
    auto F = [&](int i) { return (const float*)d_in[i]; };
    auto I = [&](int i) { return (const int*)d_in[i]; };

    const int bs = 32;
    const int nodes[9] = {6890, 3445, 1723, 862, 431, 216, 108, 54, 27};
    const int cinA[8]  = {544, 544, 544, 288, 288, 160, 160, 96};
    const int coutA[8] = {512, 512, 256, 256, 128, 128, 64, 64};

    // workspace arena (fp32 slot offsets)
    float* ws = (float*)d_ws;
    const size_t OFF_B1  = 21166080;          // x_un / prop  max 32*6890*96
    const size_t OFF_B2  = 42332160;          // t / V        max 32*6890*96
    const size_t OFF_B3  = 63498240;          // U / c3 / h   max 32*6890*64
    const size_t OFF_B4  = 77608960;          // propagate(U) max 32*6890*32
    const size_t OFF_SM  = 84664320;          // small scratch (64K floats)
    const size_t OFF_ABF = 84729856;          // bf16 A stage: 21,166,080 elems
    const size_t OFF_WBF = 95312896;          // bf16 W stage: <=278,528 elems
    float* B0 = ws;                           // h_cat        max 32*6890*96
    float* B1 = ws + OFF_B1;
    float* B2 = ws + OFF_B2;
    float* B3 = ws + OFF_B3;
    float* B4 = ws + OFF_B4;
    float* y0     = ws + OFF_SM;              // (32,63)
    float* y1     = y0 + 32 * 64;             // (32,32)
    float* catA   = y1 + 32 * 32;             // (32,160)
    float* fc1out = catA + 32 * 160;          // (32,1728) == (32,27,64)
    unsigned* abuf = (unsigned*)(ws + OFF_ABF);
    unsigned* wbuf = (unsigned*)(ws + OFF_WBF);

    auto gnrelu = [&](const float* x, float* y, const float* g, const float* b,
                      int Nn, int C) {
        groupnorm_relu<<<bs * 32, 256, 0, stream>>>(x, y, g, b, Nn, C);
    };

    // ---- pose conditioning MLP: y1 = (lrelu(pose@W1+b1))@W2+b2 ----
    dense_small<<<gsgrid(32 * 63), 256, 0, stream>>>(F(IN_POSE), F(P_PFC1W), F(P_PFC1B),
                                                     y0, 32, 126, 63, 1);
    dense_small<<<gsgrid(32 * 32), 256, 0, stream>>>(y0, F(P_PFC2W), F(P_PFC2B),
                                                     y1, 32, 63, 32, 0);

    // ---- h = lrelu([x,y1]@fc1 + b) ; reshape (32,27,64) ----
    concat_cond<<<gsgrid((size_t)bs * 160), 256, 0, stream>>>(F(IN_X), y1, catA, 1, 128, bs);
    gemm_launch<FUSE_BIAS | FUSE_LRELU>(catA, F(P_FC1W), F(P_FC1B), nullptr, fc1out,
                                        32, 1728, 160, abuf, wbuf, stream);

    // ---- conv1 (Cheb K=1): (32*27,64)@(64,512) ----
    gemm_launch<0>(fc1out, F(P_CONV1W), nullptr, nullptr, B3, 32 * 27, 512, 64,
                   abuf, wbuf, stream);
    concat_cond<<<gsgrid((size_t)bs * 27 * 544), 256, 0, stream>>>(B3, y1, B0, 27, 512, bs);

    // ---- 8 residual upsampling blocks ----
    for (int i = 0; i < 8; ++i) {
        const int lvl = 7 - i;                 // graph level / upsample index
        const int Np  = nodes[8 - i];          // source node count
        const int Nc  = nodes[lvl];            // target node count
        const int Cin = cinA[i];
        const int Co  = coutA[i];
        const int Cm  = Co >> 1;
        const int M   = bs * Nc;
        const int E   = 6 * Nc;
        const int b9  = P_BLOCKS + i * 10;
        const int* lrow = I(G_LEVELS + 3 * lvl);
        const int* lcol = I(G_LEVELS + 3 * lvl + 1);
        const float* lnorm = F(G_LEVELS + 3 * lvl + 2);
        const int* ucol = I(G_UPS + 4 * lvl + 1);
        const float* uval = F(G_UPS + 4 * lvl + 2);

        // x_un = pool(h_cat)
        pool_up<<<gsgrid((size_t)M * Cin), 256, 0, stream>>>(B0, B1, ucol, uval, Nc, Np, Cin, bs);
        // t = relu(gn1(x_un))
        gnrelu(B1, B2, F(b9 + 4), F(b9 + 5), Nc, Cin);
        // U = t @ c1 ; relu(gn2) in place
        gemm_launch<0>(B2, F(b9 + 0), nullptr, nullptr, B3, M, Cm, Cin, abuf, wbuf, stream);
        gnrelu(B3, B3, F(b9 + 6), F(b9 + 7), Nc, Cm);
        // P = propagate(U)
        zero_buf<<<gsgrid((size_t)M * Cm), 256, 0, stream>>>(B4, (size_t)M * Cm);
        scatter_prop<<<gsgrid((size_t)bs * E * Cm), 256, 0, stream>>>(B3, B4, lrow, lcol,
                                                                      lnorm, E, Nc, Cm, bs);
        // V = U@c2[0] + P@c2[1] ; relu(gn3) in place
        gemm_launch<0>(B3, F(b9 + 1), nullptr, nullptr, B2, M, Cm, Cm, abuf, wbuf, stream);
        gemm_launch<FUSE_RES>(B4, F(b9 + 1) + (size_t)Cm * Cm, nullptr, B2, B2,
                              M, Cm, Cm, abuf, wbuf, stream);
        gnrelu(B2, B2, F(b9 + 8), F(b9 + 9), Nc, Cm);
        // h = V@c3 + x_un@c4
        gemm_launch<0>(B2, F(b9 + 2), nullptr, nullptr, B3, M, Co, Cm, abuf, wbuf, stream);
        gemm_launch<FUSE_RES>(B1, F(b9 + 3), nullptr, B3, B3, M, Co, Cin, abuf, wbuf, stream);
        // h_cat = [h, y1]
        concat_cond<<<gsgrid((size_t)M * (Co + 32)), 256, 0, stream>>>(B3, y1, B0, Nc, Co, bs);
    }

    // ---- output head: Cheb K=2 (96 -> 3) on level 0 + out_bias ----
    {
        const int Nn = 6890, C = 96, E = 6 * Nn, M = bs * Nn;
        zero_buf<<<gsgrid((size_t)M * C), 256, 0, stream>>>(B1, (size_t)M * C);
        scatter_prop<<<gsgrid((size_t)bs * E * C), 256, 0, stream>>>(
            B0, B1, I(G_LEVELS), I(G_LEVELS + 1), F(G_LEVELS + 2), E, Nn, C, bs);
        out_head<<<gsgrid((size_t)M), 256, 0, stream>>>(B0, B1, F(P_CONVOUTW), F(P_OUTBIAS),
                                                        (float*)d_out, bs, Nn);
    }
}
